// FusedProjectionPlusCrossEntropyLoss_28922309771464
// MI455X (gfx1250) — matrix operand order repaired and checked by reading, not compiled, not run on hardware
//
#include <hip/hip_runtime.h>
#include <hip/hip_bf16.h>
#include <math.h>

typedef __attribute__((ext_vector_type(16))) __bf16 v16bf;
typedef __attribute__((ext_vector_type(8)))  __bf16 v8bf;
typedef __attribute__((ext_vector_type(4)))  __bf16 v4bf;
typedef __attribute__((ext_vector_type(8)))  float  v8f;

#define DIM     1024
#define VCLS    32000
#define MB      32            // tokens per workgroup (2 row-tiles of 16)
#define NRT     (MB / 16)     // row-tiles per wave
#define NTILES  (VCLS / 16)   // 2000 class tiles
#define NKB     (DIM / 32)    // 32 K-steps per class tile
#define RS      1032          // LDS row stride (bf16), padded: conflict-free b128 reads
#define IGN     (-100)
#define NWAVES  8
#define SCAL_N  256           // floats reserved for per-block partials
#define WBF_OFF 4096          // byte offset of bf16 W copy inside workspace

// ---------------- prep: divisor + zero per-block partials ----------------
__global__ void ce_prep_kernel(const long long* __restrict__ targ, int ntok,
                               float* __restrict__ scal) {
    __shared__ int cnt;
    if (threadIdx.x == 0) cnt = 0;
    __syncthreads();
    int local = 0;
    for (int i = threadIdx.x; i < ntok; i += blockDim.x)
        local += (targ[i] != (long long)IGN) ? 1 : 0;
    atomicAdd(&cnt, local);
    __syncthreads();
    for (int i = threadIdx.x; i < SCAL_N; i += blockDim.x) scal[i] = 0.0f;
    if (threadIdx.x == 0) scal[SCAL_N] = (float)cnt;   // divisor
}

// ---------------- one-time W fp32 -> bf16 (removes cvt from hot loop) ------
__global__ void wcvt_kernel(const float* __restrict__ Wf, __bf16* __restrict__ Wb) {
    const long long n4 = (long long)VCLS * DIM / 4;
    const long long stride = (long long)gridDim.x * blockDim.x;
    for (long long i = (long long)blockIdx.x * blockDim.x + threadIdx.x; i < n4; i += stride) {
        float4 f = ((const float4*)Wf)[i];
        v4bf b;
        b[0] = (__bf16)f.x; b[1] = (__bf16)f.y;
        b[2] = (__bf16)f.z; b[3] = (__bf16)f.w;
        *(v4bf*)(Wb + 4 * i) = b;
    }
}

// ---------------- main: fused GEMM + online logsumexp ----------------
template <bool PRE>
__global__ void __launch_bounds__(256)
fused_proj_ce_kernel(const float* __restrict__ x,
                     const float* __restrict__ Wf,
                     const __bf16* __restrict__ Wb,
                     const long long* __restrict__ targ,
                     float* __restrict__ scal, int ntok) {
    __shared__ __bf16 ldsA[MB * RS];          // 32 x 1024 bf16 x-block, padded rows
    __shared__ float  m_part[NWAVES][MB];
    __shared__ float  s_part[NWAVES][MB];
    __shared__ float  gt_val[MB];
    __shared__ float  loss_l[MB];
    __shared__ int    eff_t[MB];
    __shared__ int    gt_cls[MB];

    const int tid  = threadIdx.x;
    const int wave = tid >> 5;
    const int lane = tid & 31;
    const int hl   = lane >> 4;
    const int ln   = lane & 15;
    const int row0 = blockIdx.x * MB;

    // ---- effective (shifted) targets ----
    if (tid < MB) {
        const int idx   = row0 + tid;
        const int chunk = ntok / 4;            // N_LOOP_ITERS = 4
        const int pos   = idx % chunk;
        long long t1 = (idx + 1 < ntok) ? targ[idx + 1] : (long long)IGN;
        long long t2 = (idx + 2 < ntok) ? targ[idx + 2] : (long long)IGN;
        long long e  = (pos < chunk - 1) ? t1
                       : ((idx == ntok - 1) ? (long long)IGN : t2);
        eff_t[tid] = (int)e;
        int c = (int)e;
        c = c < 0 ? 0 : (c >= VCLS ? VCLS - 1 : c);
        gt_cls[tid] = c;
    }

    // ---- stage x block into LDS as bf16 ----
    for (int i = tid; i < MB * DIM; i += blockDim.x) {
        const int r = i >> 10;
        const int k = i & (DIM - 1);
        ldsA[r * RS + k] = (__bf16)x[(size_t)(row0 + r) * DIM + k];
    }
    __syncthreads();

    // ---- ground-truth logits via direct dot (keeps hot loop branch-free) ----
    if (tid < MB) {
        const int cls = gt_cls[tid];
        float d = 0.0f;
        for (int k = 0; k < DIM; ++k) {
            const float wv = PRE ? (float)Wb[(size_t)cls * DIM + k]
                                 : (float)(__bf16)Wf[(size_t)cls * DIM + k];
            d += (float)ldsA[tid * RS + k] * wv;
        }
        gt_val[tid] = d;
    }

    // ---- per-lane online softmax stats over owned columns ----
    float mrun[NRT][8], srun[NRT][8];
#pragma unroll
    for (int rt = 0; rt < NRT; ++rt)
#pragma unroll
        for (int v = 0; v < 8; ++v) { mrun[rt][v] = -INFINITY; srun[rt][v] = 0.0f; }

    const __bf16* arow0 = &ldsA[ln * RS + 16 * hl];           // row-tile 0
    const __bf16* arow1 = &ldsA[(16 + ln) * RS + 16 * hl];    // row-tile 1

    // ---- sweep class tiles: wave w owns tiles w, w+8, ... ----
    for (int j = wave; j < NTILES; j += NWAVES) {
        const int c0 = j * 16;
        const __bf16* bB = Wb + (size_t)(c0 + ln) * DIM + 16 * hl;
        const float*  bF = Wf + (size_t)(c0 + ln) * DIM + 16 * hl;

        v8f acc0 = {0.f, 0.f, 0.f, 0.f, 0.f, 0.f, 0.f, 0.f};
        v8f acc1 = {0.f, 0.f, 0.f, 0.f, 0.f, 0.f, 0.f, 0.f};

#pragma unroll 2
        for (int kb = 0; kb < NKB; ++kb) {
            // B 32x16 bf16 frag: lane holds col c0+ln, K = kb*32 + 16*hl + 0..15
            v16bf b;
            if (PRE) {
                const v8bf* bp = (const v8bf*)(bB + kb * 32);
                *((v8bf*)&b)       = bp[0];
                *(((v8bf*)&b) + 1) = bp[1];
            } else {
                const float4* wp = (const float4*)(bF + kb * 32);
                float4 f0 = wp[0], f1 = wp[1], f2 = wp[2], f3 = wp[3];
                b[0]  = (__bf16)f0.x; b[1]  = (__bf16)f0.y;
                b[2]  = (__bf16)f0.z; b[3]  = (__bf16)f0.w;
                b[4]  = (__bf16)f1.x; b[5]  = (__bf16)f1.y;
                b[6]  = (__bf16)f1.z; b[7]  = (__bf16)f1.w;
                b[8]  = (__bf16)f2.x; b[9]  = (__bf16)f2.y;
                b[10] = (__bf16)f2.z; b[11] = (__bf16)f2.w;
                b[12] = (__bf16)f3.x; b[13] = (__bf16)f3.y;
                b[14] = (__bf16)f3.z; b[15] = (__bf16)f3.w;
            }
            // A frags from LDS (row-tile 0 and 1), two b128 each
            v16bf a0, a1;
            *((v8bf*)&a0)       = *(const v8bf*)(arow0 + kb * 32);
            *(((v8bf*)&a0) + 1) = *(const v8bf*)(arow0 + kb * 32 + 8);
            *((v8bf*)&a1)       = *(const v8bf*)(arow1 + kb * 32);
            *(((v8bf*)&a1) + 1) = *(const v8bf*)(arow1 + kb * 32 + 8);

            acc0 = __builtin_amdgcn_wmma_f32_16x16x32_bf16(
                false, a0, false, b, (short)0, acc0, false, false);
            acc1 = __builtin_amdgcn_wmma_f32_16x16x32_bf16(
                false, a1, false, b, (short)0, acc1, false, false);
        }

        // online (max, sumexp) update — one column per row per tile
#pragma unroll
        for (int v = 0; v < 8; ++v) {
            {
                const float c  = acc0[v];
                const float nm = fmaxf(mrun[0][v], c);
                srun[0][v] = srun[0][v] * __expf(mrun[0][v] - nm) + __expf(c - nm);
                mrun[0][v] = nm;
            }
            {
                const float c  = acc1[v];
                const float nm = fmaxf(mrun[1][v], c);
                srun[1][v] = srun[1][v] * __expf(mrun[1][v] - nm) + __expf(c - nm);
                mrun[1][v] = nm;
            }
        }
    }

    // ---- merge (max,sum) across the 16 lanes of each half-wave ----
#pragma unroll
    for (int rt = 0; rt < NRT; ++rt) {
#pragma unroll
        for (int v = 0; v < 8; ++v) {
            float m = mrun[rt][v], s = srun[rt][v];
            for (int off = 1; off < 16; off <<= 1) {
                const float mo = __shfl_xor(m, off, 32);
                const float so = __shfl_xor(s, off, 32);
                const float nm = fmaxf(m, mo);
                s = s * __expf(m - nm) + so * __expf(mo - nm);
                m = nm;
            }
            if (ln == 0) {
                const int r = rt * 16 + hl * 8 + v;
                m_part[wave][r] = m;
                s_part[wave][r] = s;
            }
        }
    }
    __syncthreads();

    // ---- per-row finalize across the 8 waves ----
    if (tid < MB) {
        float m = m_part[0][tid];
#pragma unroll
        for (int w = 1; w < NWAVES; ++w) m = fmaxf(m, m_part[w][tid]);
        float s = 0.0f;
#pragma unroll
        for (int w = 0; w < NWAVES; ++w)
            s += s_part[w][tid] * __expf(m_part[w][tid] - m);
        const float lse = m + __logf(s);
        loss_l[tid] = (eff_t[tid] == IGN) ? 0.0f : (lse - gt_val[tid]);
    }
    __syncthreads();

    if (tid == 0) {
        float sum = 0.0f;
        for (int r = 0; r < MB; ++r) sum += loss_l[r];
        scal[blockIdx.x] = sum;          // deterministic per-block slot
    }
}

// ---------------- finalize: deterministic sum + normalize ----------------
__global__ void ce_final_kernel(const float* __restrict__ scal,
                                float* __restrict__ out, int nblk) {
    float sum = 0.0f;
    for (int i = 0; i < nblk; ++i) sum += scal[i];
    out[0] = sum / scal[SCAL_N];
}

extern "C" void kernel_launch(void* const* d_in, const int* in_sizes, int n_in,
                              void* d_out, int out_size, void* d_ws, size_t ws_size,
                              hipStream_t stream) {
    const float*     x    = (const float*)d_in[0];
    const float*     Wf   = (const float*)d_in[1];
    const long long* targ = (const long long*)d_in[2];
    float*           out  = (float*)d_out;
    float*           scal = (float*)d_ws;
    __bf16*          Wb   = (__bf16*)((char*)d_ws + WBF_OFF);

    const int ntok = in_sizes[2];          // 4096
    const int nblk = ntok / MB;            // 128

    const bool pre = ws_size >= (size_t)WBF_OFF + (size_t)VCLS * DIM * 2;

    ce_prep_kernel<<<1, 256, 0, stream>>>(targ, ntok, scal);
    if (pre) {
        wcvt_kernel<<<2048, 256, 0, stream>>>(Wf, Wb);
        fused_proj_ce_kernel<true><<<nblk, 256, 0, stream>>>(x, Wf, Wb, targ, scal, ntok);
    } else {
        fused_proj_ce_kernel<false><<<nblk, 256, 0, stream>>>(x, Wf, Wb, targ, scal, ntok);
    }
    ce_final_kernel<<<1, 1, 0, stream>>>(scal, out, nblk);
}
